// Network_conn_35957466202558
// MI455X (gfx1250) — matrix operand order repaired and checked by reading, not compiled, not run on hardware
//
#include <hip/hip_runtime.h>
#include <hip/hip_bf16.h>
#include <math.h>
#include <stdint.h>

// ---------------- model constants ----------------
#define BQ   512
#define SS   34
#define BS   (BQ * SS)      // 17408 rows
#define HIDD 256
#define FFND 512
#define VDD  512
#define NH   8
#define HSD  32
#define HVD  64
#define HGD  128
#define IND  28
#define NL   24

typedef __attribute__((ext_vector_type(16))) _Float16 v16h;
typedef __attribute__((ext_vector_type(8)))  float    v8f;

union Frag16 { float4 q[2]; v16h h; };

// ---- CDNA5 async global->LDS copies (ASYNCcnt-tracked), via inline asm ----
__device__ __forceinline__ void async_b64(unsigned lds_addr, const void* gaddr) {
    asm volatile("global_load_async_to_lds_b64 %0, %1, off"
                 :: "v"(lds_addr), "v"(gaddr) : "memory");
}
__device__ __forceinline__ void async_b128(unsigned lds_addr, const void* gaddr) {
    asm volatile("global_load_async_to_lds_b128 %0, %1, off"
                 :: "v"(lds_addr), "v"(gaddr) : "memory");
}
__device__ __forceinline__ void wait_async0() {
    asm volatile("s_wait_asynccnt 0" ::: "memory");
}
__device__ __forceinline__ unsigned lds_addr_of(const void* p) {
    // generic pointer to LDS: low 32 bits are the LDS byte offset
    return (unsigned)(uintptr_t)p;
}

// =====================================================================
// Weight conversion: gather + transpose all GEMM weights into f16 [N][K]
// Per-layer block of 786432 f16:
//   [0,393216)      QKVG^T [1536][256]
//   [393216,524288) WO^T   [256][512]
//   [524288,655360) F1^T   [512][256]
//   [655360,786432) F2^T   [256][512]
// After 24 layers: FC1^T [128][8704], FC2^T [128][128]
// =====================================================================
__global__ void wconv_kernel(const float* __restrict__ wq, const float* __restrict__ wk,
                             const float* __restrict__ wv, const float* __restrict__ wg,
                             const float* __restrict__ wo, const float* __restrict__ f1w,
                             const float* __restrict__ f2w, const float* __restrict__ fc1w,
                             const float* __restrict__ fc2w, _Float16* __restrict__ W)
{
    const long long PL  = 786432;
    const long long TOT = PL * NL + 1114112LL + 16384LL;
    for (long long gid = blockIdx.x * (long long)blockDim.x + threadIdx.x;
         gid < TOT; gid += (long long)gridDim.x * blockDim.x) {
        float val;
        if (gid < PL * NL) {
            int l = (int)(gid / PL);
            long long r = gid % PL;
            if (r < 393216) {
                int n = (int)(r / 256), d = (int)(r % 256);
                if (n < 256)      { int h = n >> 5, e = n & 31; val = wq[(((size_t)l*NH+h)*HIDD+d)*HSD + e]; }
                else if (n < 512) { int m = n-256; int h = m >> 5, e = m & 31; val = wk[(((size_t)l*NH+h)*HIDD+d)*HSD + e]; }
                else if (n < 1024){ int m = n-512; int h = m >> 6, u = m & 63; val = wv[(((size_t)l*NH+h)*HIDD+d)*HVD + u]; }
                else              { int j = n-1024; val = wg[((size_t)l*HIDD+d)*VDD + j]; }
            } else if (r < 524288) {
                long long q = r - 393216; int i = (int)(q / 512), u = (int)(q % 512);
                val = wo[((size_t)l*VDD + u)*HIDD + i];
            } else if (r < 655360) {
                long long q = r - 524288; int n = (int)(q / 256), d = (int)(q % 256);
                val = f1w[((size_t)l*HIDD + d)*FFND + n];
            } else {
                long long q = r - 655360; int i = (int)(q / 512), u = (int)(q % 512);
                val = f2w[((size_t)l*FFND + u)*HIDD + i];
            }
        } else if (gid < PL * NL + 1114112LL) {
            long long q = gid - PL * NL; int n = (int)(q / 8704), kk = (int)(q % 8704);
            val = fc1w[(size_t)kk * 128 + n];
        } else {
            long long q = gid - PL * NL - 1114112LL; int n = (int)(q / 128), kk = (int)(q % 128);
            val = fc2w[(size_t)kk * 128 + n];
        }
        W[gid] = (_Float16)val;
    }
}

// =====================================================================
// BiGRU: one block per (batch, direction, sequence); 128 threads = hidden
// =====================================================================
__global__ __launch_bounds__(128) void gru_kernel(
    const float* __restrict__ xp, const float* __restrict__ xc,
    const float* __restrict__ gp_wih, const float* __restrict__ gp_whh,
    const float* __restrict__ gp_bih, const float* __restrict__ gp_bhh,
    const float* __restrict__ gc_wih, const float* __restrict__ gc_whh,
    const float* __restrict__ gc_bih, const float* __restrict__ gc_bhh,
    float* __restrict__ X)
{
    const int b = blockIdx.x, dir = blockIdx.y, seq = blockIdx.z;
    const int j = threadIdx.x;
    const int T = seq ? 20 : 14;
    const int sBase = seq ? 14 : 0;
    const float* x   = seq ? xc : xp;
    const float* wih = (seq ? gc_wih : gp_wih) + (size_t)dir * 384 * IND;
    const float* whh = (seq ? gc_whh : gp_whh) + (size_t)dir * 384 * HGD;
    const float* bih = (seq ? gc_bih : gp_bih) + dir * 384;
    const float* bhh = (seq ? gc_bhh : gp_bhh) + dir * 384;

    __shared__ float hs[HGD];
    __shared__ float xs[IND];
    hs[j] = 0.0f;
    __syncthreads();
    for (int step = 0; step < T; ++step) {
        const int tt = dir ? (T - 1 - step) : step;
        if (j < IND) xs[j] = x[((size_t)b * T + tt) * IND + j];
        __syncthreads();
        float xr = bih[j], xz = bih[128 + j], xn = bih[256 + j];
        for (int i = 0; i < IND; ++i) {
            const float xi = xs[i];
            xr += xi * wih[(size_t)j * IND + i];
            xz += xi * wih[(size_t)(128 + j) * IND + i];
            xn += xi * wih[(size_t)(256 + j) * IND + i];
        }
        float hr = bhh[j], hz = bhh[128 + j], hn = bhh[256 + j];
        for (int m = 0; m < HGD; ++m) {
            const float hm = hs[m];
            hr += hm * whh[(size_t)j * HGD + m];
            hz += hm * whh[(size_t)(128 + j) * HGD + m];
            hn += hm * whh[(size_t)(256 + j) * HGD + m];
        }
        const float h0 = hs[j];
        const float r = 1.0f / (1.0f + expf(-(xr + hr)));
        const float z = 1.0f / (1.0f + expf(-(xz + hz)));
        const float n = tanhf(xn + r * hn);
        const float hnew = (1.0f - z) * n + z * h0;
        __syncthreads();
        hs[j] = hnew;
        X[((size_t)b * SS + sBase + tt) * HIDD + dir * HGD + j] = hnew;
        __syncthreads();
    }
}

// =====================================================================
// LayerNorm(256) -> f16 ; one wave per row, 8 waves per block
// =====================================================================
__global__ __launch_bounds__(256) void ln_f16_kernel(
    const float* __restrict__ X, const float* __restrict__ g,
    const float* __restrict__ bta, _Float16* __restrict__ H, int rows)
{
    const int wave = threadIdx.x >> 5, lane = threadIdx.x & 31;
    const int row = blockIdx.x * 8 + wave;
    if (row >= rows) return;
    const float* xr = X + (size_t)row * HIDD;
    float v[8], s = 0.0f, s2 = 0.0f;
    for (int i = 0; i < 8; ++i) { v[i] = xr[lane + 32 * i]; s += v[i]; s2 += v[i] * v[i]; }
    for (int off = 16; off; off >>= 1) { s += __shfl_xor(s, off, 32); s2 += __shfl_xor(s2, off, 32); }
    const float mu = s * (1.0f / 256.0f);
    const float var = s2 * (1.0f / 256.0f) - mu * mu;
    const float rs = rsqrtf(var + 1e-5f);
    _Float16* hr = H + (size_t)row * HIDD;
    for (int i = 0; i < 8; ++i) {
        const int c = lane + 32 * i;
        hr[c] = (_Float16)((v[i] - mu) * rs * g[c] + bta[c]);
    }
}

// =====================================================================
// WMMA f16 GEMM:  out = act(A[M,K] @ WT[N,K]^T + bias) (+res)
// 8 waves/block; macro-tile 32(M) x 128(N); each wave does 16x32 (2 accs).
// K-step 32, double-buffered LDS filled by GLOBAL_LOAD_ASYNC_TO_LDS
// (ASYNCcnt), so stage i+1 loads overlap stage i WMMAs.
// Fragments (16-bit A 16x32, B 32x16) use contiguous k: lane half h holds k 16h..16h+15.
// act: 0=none 1=gelu(exact) 2=relu
// =====================================================================
__global__ __launch_bounds__(256) void gemm16(
    const _Float16* __restrict__ A, const _Float16* __restrict__ WT,
    const float* __restrict__ bias, const float* __restrict__ res,
    float* __restrict__ outF, _Float16* __restrict__ outH,
    int M, int N, int K, int act)
{
    __shared__ alignas(16) _Float16 lA[2][32][32];    // 2 x 2KB
    __shared__ alignas(16) _Float16 lB[2][128][32];   // 2 x 8KB
    const int t = threadIdx.x;
    const int lane = t & 31, wave = t >> 5;
    const int m0 = blockIdx.y * 32, n0 = blockIdx.x * 128;
    const int wm = (wave >> 2) * 16, wn = (wave & 3) * 32;

    // per-thread staging slots
    const int ar = t >> 3, ac = (t & 7) * 4;   // A: 32 rows, 8 thr/row, 4 f16 (8B) each
    const int br = t >> 1, bc = (t & 1) * 16;  // B: 128 rows, 2 thr/row, 16 f16 (32B) each

    const _Float16* gA = &A [(size_t)(m0 + ar) * K + ac];
    const _Float16* gB = &WT[(size_t)(n0 + br) * K + bc];

    auto issue = [&](int buf, int k0) {
        async_b64 (lds_addr_of(&lA[buf][ar][ac]),      gA + k0);
        async_b128(lds_addr_of(&lB[buf][br][bc]),      gB + k0);
        async_b128(lds_addr_of(&lB[buf][br][bc] + 8),  gB + k0 + 8);
    };

    v8f acc0 = {}, acc1 = {};
    const int NSTG = K / 32;
    issue(0, 0);
    for (int s = 0; s < NSTG; ++s) {
        wait_async0();            // this wave's stage-s tile is in LDS
        __syncthreads();          // everyone's stage-s tile is in LDS; stage s-1 reads done
        if (s + 1 < NSTG) issue((s + 1) & 1, (s + 1) * 32);  // overlap with compute
        const int buf = s & 1;
        Frag16 fa, fb0, fb1;
        { // A frag: lane<16 -> K[0..15]; lane>=16 -> K[16..31]
            const int row = wm + (lane & 15);
            const int ka = (lane >> 4) * 16;
            fa.q[0] = *(const float4*)&lA[buf][row][ka];
            fa.q[1] = *(const float4*)&lA[buf][row][ka + 8];
        }
        { // B frags: lane<16 -> K[0..15]; lane>=16 -> K[16..31]
            const int col = wn + (lane & 15);
            const int kb = (lane >> 4) * 16;
            fb0.q[0] = *(const float4*)&lB[buf][col][kb];
            fb0.q[1] = *(const float4*)&lB[buf][col][kb + 8];
            fb1.q[0] = *(const float4*)&lB[buf][col + 16][kb];
            fb1.q[1] = *(const float4*)&lB[buf][col + 16][kb + 8];
        }
        acc0 = __builtin_amdgcn_wmma_f32_16x16x32_f16(false, fa.h, false, fb0.h,
                                                      (short)0, acc0, false, false);
        acc1 = __builtin_amdgcn_wmma_f32_16x16x32_f16(false, fa.h, false, fb1.h,
                                                      (short)0, acc1, false, false);
    }
    // epilogue: two 16x16 tiles per wave
    for (int c = 0; c < 2; ++c) {
        const v8f acc = c ? acc1 : acc0;
        const int gc = n0 + wn + 16 * c + (lane & 15);
        const float bv = bias ? bias[gc] : 0.0f;
        for (int r = 0; r < 8; ++r) {
            const int gr = m0 + wm + r + ((lane >> 4) * 8);
            float v = acc[r] + bv;
            if (act == 1)      v = 0.5f * v * (1.0f + erff(v * 0.7071067811865476f));
            else if (act == 2) v = fmaxf(v, 0.0f);
            if (res)  v += res[(size_t)gr * N + gc];
            if (outF) outF[(size_t)gr * N + gc] = v;
            if (outH) outH[(size_t)gr * N + gc] = (_Float16)v;
        }
    }
}

// =====================================================================
// xPos rotary on q,k (in-place on f16 QKVG buffer)
// =====================================================================
__global__ void xpos_kernel(_Float16* __restrict__ QK)
{
    const int total = BQ * SS * NH * 16;
    const int idx = blockIdx.x * blockDim.x + threadIdx.x;
    if (idx >= total) return;
    const int i   = idx & 15;
    const int h   = (idx >> 4) & 7;
    const int row = idx >> 7;        // b*S+s
    const int pos = row % SS;
    const float scale = (2.0f * (float)i + 12.8f) / 44.8f;
    const float sm  = powf(scale, (float)pos * (1.0f / 512.0f));
    const float inv = powf(10000.0f, -((float)i) / 16.0f);
    const float ang = (float)pos * inv;
    const float sn = sinf(ang), cs = cosf(ang);
    _Float16* q = QK + (size_t)row * 1536 + h * HSD + 2 * i;
    _Float16* k = q + 256;
    {
        const float q0 = (float)q[0], q1 = (float)q[1];
        const float s = sn * sm, c = cs * sm;
        q[0] = (_Float16)(-q1 * s + q0 * c);
        q[1] = (_Float16)( q0 * s + q1 * c);
    }
    {
        const float ism = 1.0f / sm;
        const float k0 = (float)k[0], k1 = (float)k[1];
        const float s = sn * ism, c = cs * ism;
        k[0] = (_Float16)(-k1 * s + k0 * c);
        k[1] = (_Float16)( k0 * s + k1 * c);
    }
}

// =====================================================================
// Retention attention per (head, batch): y = (q k^T * D) v   (S=34)
// =====================================================================
__global__ __launch_bounds__(256) void attn_kernel(
    const _Float16* __restrict__ QKVG, float* __restrict__ Y)
{
    const int h = blockIdx.x, b = blockIdx.y;
    __shared__ float qs[SS][HSD], ks[SS][HSD], vs[SS][HVD], at[SS][SS];
    const int t = threadIdx.x;
    const _Float16* base = QKVG + (size_t)b * SS * 1536;
    for (int idx = t; idx < SS * HSD; idx += 256) {
        const int s = idx >> 5, e = idx & 31;
        qs[s][e] = (float)base[(size_t)s * 1536 + h * HSD + e];
        ks[s][e] = (float)base[(size_t)s * 1536 + 256 + h * HSD + e];
    }
    for (int idx = t; idx < SS * HVD; idx += 256) {
        const int s = idx >> 6, u = idx & 63;
        vs[s][u] = (float)base[(size_t)s * 1536 + 512 + h * HVD + u];
    }
    __syncthreads();
    const float l32 = -3.4657359027997265f, l512 = -6.238324625039508f;
    const float gamma = 1.0f - expf(l32 + (l512 - l32) * ((float)h / 7.0f));
    const float lg = logf(gamma);
    for (int idx = t; idx < SS * SS; idx += 256) {
        const int si = idx / SS, ti = idx % SS;
        float a = 0.0f;
        if (si >= ti) {
            float d = 0.0f;
            for (int e = 0; e < HSD; ++e) d += qs[si][e] * ks[ti][e];
            a = d * expf(lg * (float)(si - ti));
        }
        at[si][ti] = a;
    }
    __syncthreads();
    for (int idx = t; idx < SS * HVD; idx += 256) {
        const int si = idx >> 6, u = idx & 63;
        float a = 0.0f;
        for (int tt = 0; tt <= si; ++tt) a += at[si][tt] * vs[tt][u];
        Y[((size_t)b * SS + si) * VDD + h * HVD + u] = a;
    }
}

// =====================================================================
// GroupNorm (per-head, 64) + affine + SiLU gate -> f16
// =====================================================================
__global__ __launch_bounds__(256) void gnorm_gate_kernel(
    const float* __restrict__ Y, const _Float16* __restrict__ QKVG,
    const float* __restrict__ gng, const float* __restrict__ gnb,
    _Float16* __restrict__ YG)
{
    const int row = blockIdx.x;
    const int head = threadIdx.x >> 5, lane = threadIdx.x & 31;
    const float* yr = Y + (size_t)row * VDD + head * HVD;
    const float y0 = yr[lane], y1 = yr[lane + 32];
    float s = y0 + y1, s2 = y0 * y0 + y1 * y1;
    for (int off = 16; off; off >>= 1) { s += __shfl_xor(s, off, 32); s2 += __shfl_xor(s2, off, 32); }
    const float mu = s * (1.0f / 64.0f);
    const float var = s2 * (1.0f / 64.0f) - mu * mu;
    const float rs = rsqrtf(var + 1e-5f);
    const _Float16* gp = QKVG + (size_t)row * 1536 + 1024 + head * HVD;
    _Float16* og = YG + (size_t)row * VDD + head * HVD;
    const int jb = head * HVD;
    {
        const float yn = (y0 - mu) * rs * gng[jb + lane] + gnb[jb + lane];
        const float gv = (float)gp[lane];
        og[lane] = (_Float16)(gv / (1.0f + expf(-gv)) * yn);
    }
    {
        const int u = lane + 32;
        const float yn = (y1 - mu) * rs * gng[jb + u] + gnb[jb + u];
        const float gv = (float)gp[u];
        og[u] = (_Float16)(gv / (1.0f + expf(-gv)) * yn);
    }
}

// =====================================================================
// f32 -> f16 elementwise
// =====================================================================
__global__ void f2h_kernel(const float* __restrict__ in, _Float16* __restrict__ out, long long n)
{
    const long long i = blockIdx.x * (long long)blockDim.x + threadIdx.x;
    if (i < n) out[i] = (_Float16)in[i];
}

// =====================================================================
// Final fc3 + sigmoid
// =====================================================================
__global__ void fc3_kernel(const float* __restrict__ z2, const float* __restrict__ w,
                           const float* __restrict__ b, float* __restrict__ out)
{
    const int i = blockIdx.x * blockDim.x + threadIdx.x;
    if (i >= BQ) return;
    float a = b[0];
    for (int k = 0; k < 128; ++k) a += z2[(size_t)i * 128 + k] * w[k];
    out[i] = 1.0f / (1.0f + expf(-a));
}

// =====================================================================
extern "C" void kernel_launch(void* const* d_in, const int* in_sizes, int n_in,
                              void* d_out, int out_size, void* d_ws, size_t ws_size,
                              hipStream_t stream) {
    (void)in_sizes; (void)n_in; (void)out_size; (void)ws_size;
    const float* peps3 = (const float*)d_in[0];
    const float* cdr3  = (const float*)d_in[1];
    const float* gp_wih = (const float*)d_in[2];
    const float* gp_whh = (const float*)d_in[3];
    const float* gp_bih = (const float*)d_in[4];
    const float* gp_bhh = (const float*)d_in[5];
    const float* gc_wih = (const float*)d_in[6];
    const float* gc_whh = (const float*)d_in[7];
    const float* gc_bih = (const float*)d_in[8];
    const float* gc_bhh = (const float*)d_in[9];
    const float* wq   = (const float*)d_in[10];
    const float* wk   = (const float*)d_in[11];
    const float* wv   = (const float*)d_in[12];
    const float* wg   = (const float*)d_in[13];
    const float* wo   = (const float*)d_in[14];
    const float* gng  = (const float*)d_in[15];
    const float* gnb  = (const float*)d_in[16];
    const float* ln1g = (const float*)d_in[17];
    const float* ln1b = (const float*)d_in[18];
    const float* ln2g = (const float*)d_in[19];
    const float* ln2b = (const float*)d_in[20];
    const float* f1w  = (const float*)d_in[21];
    const float* f1b  = (const float*)d_in[22];
    const float* f2w  = (const float*)d_in[23];
    const float* f2b  = (const float*)d_in[24];
    const float* fc1w = (const float*)d_in[25];
    const float* fc1b = (const float*)d_in[26];
    const float* fc2w = (const float*)d_in[27];
    const float* fc2b = (const float*)d_in[28];
    const float* fc3w = (const float*)d_in[29];
    const float* fc3b = (const float*)d_in[30];

    // ---- workspace carve-out (offsets in bytes, 256-aligned) ----
    char* ws = (char*)d_ws;
    float*     X    = (float*)    (ws + 0);           // BS*256 f32   17.8MB
    float*     RET  = (float*)    (ws + 17825792);    // BS*256 f32   17.8MB
    float*     Yb   = (float*)    (ws + 35651584);    // BS*512 f32   35.6MB
    _Float16*  H16  = (_Float16*) (ws + 71303168);    // BS*256 f16    8.9MB
    _Float16*  QKVG = (_Float16*) (ws + 80216064);    // BS*1536 f16  53.5MB
    _Float16*  YG   = (_Float16*) (ws + 133693440);   // BS*512 f16   17.8MB (also FFN f16)
    _Float16*  W16  = (_Float16*) (ws + 151519232);   // 20,004,864 f16 40.0MB
    _Float16*  Z1   = (_Float16*) (ws + 191528960);   // 512*128 f16
    float*     Z2   = (float*)    (ws + 191660032);   // 512*128 f32

    // 1) convert all weights -> f16 transposed
    wconv_kernel<<<2048, 256, 0, stream>>>(wq, wk, wv, wg, wo, f1w, f2w, fc1w, fc2w, W16);

    // 2) BiGRU encoders -> X (B, S, 256)
    gru_kernel<<<dim3(BQ, 2, 2), 128, 0, stream>>>(peps3, cdr3,
        gp_wih, gp_whh, gp_bih, gp_bhh, gc_wih, gc_whh, gc_bih, gc_bhh, X);

    // 3) 24 retention layers
    for (int l = 0; l < NL; ++l) {
        const size_t WL = (size_t)l * 786432;
        ln_f16_kernel<<<BS / 8, 256, 0, stream>>>(X, ln1g + l * HIDD, ln1b + l * HIDD, H16, BS);
        // fused q,k,v,g projection: [BS,256] x [256,1536]
        gemm16<<<dim3(1536 / 128, BS / 32), 256, 0, stream>>>(
            H16, W16 + WL, nullptr, nullptr, nullptr, QKVG, BS, 1536, 256, 0);
        xpos_kernel<<<(BQ * SS * NH * 16 + 255) / 256, 256, 0, stream>>>(QKVG);
        attn_kernel<<<dim3(NH, BQ), 256, 0, stream>>>(QKVG, Yb);
        gnorm_gate_kernel<<<BS, 256, 0, stream>>>(Yb, QKVG, gng + l * VDD, gnb + l * VDD, YG);
        // output projection + residual: RET = YG @ Wo + X
        gemm16<<<dim3(256 / 128, BS / 32), 256, 0, stream>>>(
            YG, W16 + WL + 393216, nullptr, X, RET, nullptr, BS, 256, 512, 0);
        ln_f16_kernel<<<BS / 8, 256, 0, stream>>>(RET, ln2g + l * HIDD, ln2b + l * HIDD, H16, BS);
        // FFN up + exact gelu -> f16 (reuse YG buffer)
        gemm16<<<dim3(512 / 128, BS / 32), 256, 0, stream>>>(
            H16, W16 + WL + 524288, f1b + l * FFND, nullptr, nullptr, YG, BS, 512, 256, 1);
        // FFN down + bias + residual: X = YG @ F2 + b + RET
        gemm16<<<dim3(256 / 128, BS / 32), 256, 0, stream>>>(
            YG, W16 + WL + 655360, f2b + l * HIDD, RET, X, nullptr, BS, 256, 512, 0);
    }

    // 4) MLP head: z = X reshaped (512, 8704)
    f2h_kernel<<<(4456448 + 255) / 256, 256, 0, stream>>>(X, H16, 4456448LL);
    gemm16<<<dim3(128 / 128, BQ / 32), 256, 0, stream>>>(
        H16, W16 + 18874368, fc1b, nullptr, nullptr, Z1, BQ, 128, 8704, 2);
    gemm16<<<dim3(128 / 128, BQ / 32), 256, 0, stream>>>(
        Z1, W16 + 18874368 + 1114112, fc2b, nullptr, Z2, nullptr, BQ, 128, 128, 2);
    fc3_kernel<<<2, 256, 0, stream>>>(Z2, fc3w, fc3b, (float*)d_out);
}
